// ASSENT_70746701300309
// MI455X (gfx1250) — matrix operand order repaired and checked by reading, not compiled, not run on hardware
//
#include <hip/hip_runtime.h>
#include <hip/hip_bf16.h>

// ---------------------------------------------------------------------------
// Hetero-GraphSAGE forward for MI455X (gfx1250, wave32).
//   - dense 64x64 linears    -> v_wmma_f32_16x16x32_bf16 (f32 accumulate)
//   - scatter-mean to AP     -> LDS-replicated accumulator (ds_add_f32), 1 flush/block
//   - scatter-mean to user/t -> global_atomic_add_f32 (L2-resident, 31MB << 192MB L2)
//   - AP combine fuses both reverse relations (8 WMMAs, no output re-read)
// ---------------------------------------------------------------------------

#define HDIM 64
#define NEG_SLOPE 0.01f

static constexpr int N_AP = 512, N_USER = 100000, N_TARGET = 20000;
static constexpr int E_SERVES = 2000000, E_SENSES = 500000;

typedef __attribute__((ext_vector_type(16))) __bf16 bf16x16;
typedef __attribute__((ext_vector_type(8)))  float  v8f;

// ---------------- projection: out[n,o] = b[o] + sum_k x[n,k] W[o,k] ----------
__global__ void proj_kernel(const float* __restrict__ x, const float* __restrict__ W,
                            const float* __restrict__ b, float* __restrict__ out,
                            int N, int Din) {
    int i = blockIdx.x * blockDim.x + threadIdx.x;
    if (i >= N * HDIM) return;
    int n = i >> 6, o = i & 63;
    const float* xr = x + (size_t)n * Din;
    const float* wr = W + (size_t)o * Din;
    float acc = b[o];
    for (int k = 0; k < Din; ++k) acc += xr[k] * wr[k];
    out[i] = acc;
}

// -------- scatter-add into large dst sets (user/target): global atomics ------
__global__ void scatter_add_kernel(const float* __restrict__ feat,
                                   const int* __restrict__ src, const int* __restrict__ dst,
                                   float* __restrict__ sums, float* __restrict__ cnt, int E) {
    long long t = (long long)blockIdx.x * blockDim.x + threadIdx.x;
    long long e = t >> 4;                 // 16 lanes per edge, float4 per lane
    if (e >= E) return;
    int lane = (int)(t & 15);
    int s = src[e], d = dst[e];
    float4 v = ((const float4*)(feat + (size_t)s * HDIM))[lane];
    float* o = sums + (size_t)d * HDIM + lane * 4;
    atomicAdd(o + 0, v.x); atomicAdd(o + 1, v.y);
    atomicAdd(o + 2, v.z); atomicAdd(o + 3, v.w);
    if (lane == 0) atomicAdd(cnt + d, 1.0f);
}

// -------- scatter-add into AP (512 nodes): LDS-replicated accumulator --------
// 512*64 fp32 = 128KB fits in the 320KB WGP LDS -> ds_add_f32, one global flush
__global__ void scatter_add_ap_kernel(const float* __restrict__ feat,
                                      const int* __restrict__ gidx,   // gather index (src feat)
                                      const int* __restrict__ aidx,   // AP dst index
                                      float* __restrict__ sums, float* __restrict__ cnt, int E) {
    __shared__ float lsum[N_AP * HDIM];
    __shared__ float lcnt[N_AP];
    for (int i = threadIdx.x; i < N_AP * HDIM; i += blockDim.x) lsum[i] = 0.f;
    for (int i = threadIdx.x; i < N_AP; i += blockDim.x) lcnt[i] = 0.f;
    __syncthreads();
    long long total  = (long long)E * 16;
    long long stride = (long long)gridDim.x * blockDim.x;
    for (long long t = (long long)blockIdx.x * blockDim.x + threadIdx.x; t < total; t += stride) {
        long long e = t >> 4;
        int lane = (int)(t & 15);
        int g = gidx[e], a = aidx[e];
        float4 v = ((const float4*)(feat + (size_t)g * HDIM))[lane];
        float* o = lsum + a * HDIM + lane * 4;
        atomicAdd(o + 0, v.x); atomicAdd(o + 1, v.y);
        atomicAdd(o + 2, v.z); atomicAdd(o + 3, v.w);
        if (lane == 0) atomicAdd(lcnt + a, 1.0f);
    }
    __syncthreads();
    for (int i = threadIdx.x; i < N_AP * HDIM; i += blockDim.x)
        if (lsum[i] != 0.f) atomicAdd(sums + i, lsum[i]);
    for (int i = threadIdx.x; i < N_AP; i += blockDim.x)
        if (lcnt[i] != 0.f) atomicAdd(cnt + i, lcnt[i]);
}

// ---------------- sums -> means (divide by max(count,1)) ---------------------
__global__ void mean_div_kernel(float* __restrict__ sums, const float* __restrict__ cnt, int N) {
    int i = blockIdx.x * blockDim.x + threadIdx.x;
    if (i >= N * HDIM) return;
    float c = cnt[i >> 6];
    sums[i] = sums[i] / fmaxf(c, 1.0f);
}

// ---------------- WMMA fragment loaders (ISA 7.12.2 layouts) -----------------
// A (16x32 bf16): lane L -> row = L&15; half h=L>>4; VGPR v holds K pair:
//   K0 = (v<4 ? 0 : 16) + 8*h + 2*(v&3)
__device__ inline bf16x16 load_A_frag(const float* __restrict__ feat, int row, int kchunk, int lane) {
    int h = lane >> 4;
    const float* rp = feat + (size_t)row * HDIM + kchunk * 32;
    bf16x16 a;
#pragma unroll
    for (int v = 0; v < 8; ++v) {
        int k0 = ((v < 4) ? 0 : 16) + h * 8 + (v & 3) * 2;
        a[2 * v]     = (__bf16)rp[k0];
        a[2 * v + 1] = (__bf16)rp[k0 + 1];
    }
    return a;
}
// B (32x16 bf16), B[k][o] = W[o*64+k]: lane L -> col o = obase + (L&15);
//   half h; VGPR v holds K pair K0 = (v<4 ? 0 : 16) + 8*h + 2*(v&3)
__device__ inline bf16x16 load_B_frag(const float* __restrict__ W, int obase, int kchunk, int lane) {
    int n = lane & 15, h = lane >> 4;
    const float* wp = W + (size_t)(obase + n) * HDIM + kchunk * 32;
    bf16x16 b;
#pragma unroll
    for (int v = 0; v < 8; ++v) {
        int k0 = ((v < 4) ? 0 : 16) + h * 8 + (v & 3) * 2;
        b[2 * v]     = (__bf16)wp[k0];
        b[2 * v + 1] = (__bf16)wp[k0 + 1];
    }
    return b;
}

__device__ inline v8f wmma_bf16(bf16x16 a, bf16x16 b, v8f c) {
    return __builtin_amdgcn_wmma_f32_16x16x32_bf16(false, a, false, b, (short)0, c, false, false);
}

// ---- SAGE combine: out = lrelu( M@Wl^T + bl + X@Wr^T ) ----------------------
// 128 threads = 4 waves; each wave owns 16 rows x full 64 outputs (4 tiles).
// 16 x v_wmma_f32_16x16x32_bf16 per wave. Weights staged once per block in LDS.
__global__ void sage_wmma_kernel(const float* __restrict__ M, const float* __restrict__ X,
                                 const float* __restrict__ Wl, const float* __restrict__ bl,
                                 const float* __restrict__ Wr,
                                 float* __restrict__ out, int N) {
    __shared__ float sWl[HDIM * HDIM];
    __shared__ float sWr[HDIM * HDIM];
    for (int i = threadIdx.x; i < HDIM * HDIM; i += blockDim.x) { sWl[i] = Wl[i]; sWr[i] = Wr[i]; }
    __syncthreads();

    int lane = threadIdx.x & 31;
    int wave = threadIdx.x >> 5;
    int tile = (blockIdx.x * (blockDim.x >> 5) + wave) * 16;
    if (tile >= N) return;                      // wave-uniform: EXEC stays all-1s

    int row = tile + (lane & 15);
    bf16x16 aM0 = load_A_frag(M, row, 0, lane);
    bf16x16 aM1 = load_A_frag(M, row, 1, lane);
    bf16x16 aX0 = load_A_frag(X, row, 0, lane);
    bf16x16 aX1 = load_A_frag(X, row, 1, lane);

    int n = lane & 15, h = lane >> 4;
#pragma unroll
    for (int ot = 0; ot < 4; ++ot) {
        int obase = ot * 16;
        v8f acc = {};
        acc = wmma_bf16(aM0, load_B_frag(sWl, obase, 0, lane), acc);
        acc = wmma_bf16(aM1, load_B_frag(sWl, obase, 1, lane), acc);
        acc = wmma_bf16(aX0, load_B_frag(sWr, obase, 0, lane), acc);
        acc = wmma_bf16(aX1, load_B_frag(sWr, obase, 1, lane), acc);
        float bias = bl[obase + n];
#pragma unroll
        for (int v = 0; v < 8; ++v) {
            int m = tile + v + 8 * h;           // C/D layout: row = vgpr + 8*lanehalf
            float val = acc[v] + bias;
            val = (val >= 0.f) ? val : NEG_SLOPE * val;
            out[(size_t)m * HDIM + obase + n] = val;
        }
    }
}

// ---- fused AP combine: out = lrelu( M1@Wl1^T+bl1 + X@Wr1^T
//                                   + M2@Wl2^T+bl2 + X@Wr2^T ) ----------------
// Both reverse relations in one pass: 8 WMMAs per tile, no output re-read,
// no branches. 4 weight matrices staged in 64KB LDS.
__global__ void ap_combine_wmma_kernel(const float* __restrict__ M1, const float* __restrict__ M2,
                                       const float* __restrict__ X,
                                       const float* __restrict__ Wl1, const float* __restrict__ bl1,
                                       const float* __restrict__ Wr1,
                                       const float* __restrict__ Wl2, const float* __restrict__ bl2,
                                       const float* __restrict__ Wr2,
                                       float* __restrict__ out, int N) {
    __shared__ float sWl1[HDIM * HDIM], sWr1[HDIM * HDIM];
    __shared__ float sWl2[HDIM * HDIM], sWr2[HDIM * HDIM];
    for (int i = threadIdx.x; i < HDIM * HDIM; i += blockDim.x) {
        sWl1[i] = Wl1[i]; sWr1[i] = Wr1[i];
        sWl2[i] = Wl2[i]; sWr2[i] = Wr2[i];
    }
    __syncthreads();

    int lane = threadIdx.x & 31;
    int wave = threadIdx.x >> 5;
    int tile = (blockIdx.x * (blockDim.x >> 5) + wave) * 16;
    if (tile >= N) return;

    int row = tile + (lane & 15);
    bf16x16 a10 = load_A_frag(M1, row, 0, lane);
    bf16x16 a11 = load_A_frag(M1, row, 1, lane);
    bf16x16 a20 = load_A_frag(M2, row, 0, lane);
    bf16x16 a21 = load_A_frag(M2, row, 1, lane);
    bf16x16 ax0 = load_A_frag(X,  row, 0, lane);
    bf16x16 ax1 = load_A_frag(X,  row, 1, lane);

    int n = lane & 15, h = lane >> 4;
#pragma unroll
    for (int ot = 0; ot < 4; ++ot) {
        int obase = ot * 16;
        v8f acc = {};
        acc = wmma_bf16(a10, load_B_frag(sWl1, obase, 0, lane), acc);
        acc = wmma_bf16(a11, load_B_frag(sWl1, obase, 1, lane), acc);
        acc = wmma_bf16(ax0, load_B_frag(sWr1, obase, 0, lane), acc);
        acc = wmma_bf16(ax1, load_B_frag(sWr1, obase, 1, lane), acc);
        acc = wmma_bf16(a20, load_B_frag(sWl2, obase, 0, lane), acc);
        acc = wmma_bf16(a21, load_B_frag(sWl2, obase, 1, lane), acc);
        acc = wmma_bf16(ax0, load_B_frag(sWr2, obase, 0, lane), acc);
        acc = wmma_bf16(ax1, load_B_frag(sWr2, obase, 1, lane), acc);
        float bias = bl1[obase + n] + bl2[obase + n];
#pragma unroll
        for (int v = 0; v < 8; ++v) {
            int m = tile + v + 8 * h;
            float val = acc[v] + bias;
            val = (val >= 0.f) ? val : NEG_SLOPE * val;
            out[(size_t)m * HDIM + obase + n] = val;
        }
    }
}

// ---------------- node head: out[n] = X[n,:]·W + b ---------------------------
__global__ void node_head_kernel(const float* __restrict__ X, const float* __restrict__ W,
                                 const float* __restrict__ b, float* __restrict__ out, int N) {
    long long t = (long long)blockIdx.x * blockDim.x + threadIdx.x;
    long long node = t >> 4;
    if (node >= N) return;
    int lane = (int)(t & 15);
    float4 xv = ((const float4*)(X + (size_t)node * HDIM))[lane];
    float4 wv = ((const float4*)W)[lane];
    float s = xv.x * wv.x + xv.y * wv.y + xv.z * wv.z + xv.w * wv.w;
    for (int off = 8; off >= 1; off >>= 1) s += __shfl_xor(s, off, 32);
    if (lane == 0) out[node] = s + b[0];
}

// ------------- edge head: out[e] = concat(Xa[src],Xd[dst])·W + b -------------
__global__ void edge_head1_kernel(const float* __restrict__ Xa, const float* __restrict__ Xd,
                                  const int* __restrict__ src, const int* __restrict__ dst,
                                  const float* __restrict__ W, const float* __restrict__ b,
                                  float* __restrict__ out, int E) {
    long long t = (long long)blockIdx.x * blockDim.x + threadIdx.x;
    long long e = t >> 5;                 // 32 lanes per edge (2H=128 floats)
    if (e >= E) return;
    int lane = (int)(t & 31);
    int s = src[e], d = dst[e];
    float4 v = (lane < 16) ? ((const float4*)(Xa + (size_t)s * HDIM))[lane]
                           : ((const float4*)(Xd + (size_t)d * HDIM))[lane & 15];
    float4 wv = ((const float4*)W)[lane];
    float a = v.x * wv.x + v.y * wv.y + v.z * wv.z + v.w * wv.w;
    for (int off = 16; off >= 1; off >>= 1) a += __shfl_xor(a, off, 32);
    if (lane == 0) out[e] = a + b[0];
}

// fused two-head variant (ytx / yrx share the edge-feature reads)
__global__ void edge_head2_kernel(const float* __restrict__ Xa, const float* __restrict__ Xd,
                                  const int* __restrict__ src, const int* __restrict__ dst,
                                  const float* __restrict__ W1, const float* __restrict__ b1,
                                  const float* __restrict__ W2, const float* __restrict__ b2,
                                  float* __restrict__ out1, float* __restrict__ out2, int E) {
    long long t = (long long)blockIdx.x * blockDim.x + threadIdx.x;
    long long e = t >> 5;
    if (e >= E) return;
    int lane = (int)(t & 31);
    int s = src[e], d = dst[e];
    float4 v = (lane < 16) ? ((const float4*)(Xa + (size_t)s * HDIM))[lane]
                           : ((const float4*)(Xd + (size_t)d * HDIM))[lane & 15];
    float4 w1 = ((const float4*)W1)[lane];
    float4 w2 = ((const float4*)W2)[lane];
    float a1 = v.x * w1.x + v.y * w1.y + v.z * w1.z + v.w * w1.w;
    float a2 = v.x * w2.x + v.y * w2.y + v.z * w2.z + v.w * w2.w;
    for (int off = 16; off >= 1; off >>= 1) {
        a1 += __shfl_xor(a1, off, 32);
        a2 += __shfl_xor(a2, off, 32);
    }
    if (lane == 0) { out1[e] = a1 + b1[0]; out2[e] = a2 + b2[0]; }
}

// ---------------------------------------------------------------------------
extern "C" void kernel_launch(void* const* d_in, const int* in_sizes, int n_in,
                              void* d_out, int out_size, void* d_ws, size_t ws_size,
                              hipStream_t stream) {
    const float* x_ap       = (const float*)d_in[0];
    const float* x_user     = (const float*)d_in[1];
    const float* x_target   = (const float*)d_in[2];
    const int*   serves_src = (const int*)d_in[3];
    const int*   serves_dst = (const int*)d_in[4];
    const int*   senses_src = (const int*)d_in[5];
    const int*   senses_dst = (const int*)d_in[6];
    auto P = [&](int i) { return (const float*)d_in[i]; };
    // param flattening (dict order): 7..12 proj, 13..24 conv1, 25..36 conv2, 37..46 heads

    // -------- workspace carve-out (~93 MB total) --------
    float* ws = (float*)d_ws;
    size_t off = 0;
    auto alloc = [&](size_t nfloats) { float* p = ws + off; off += nfloats; return p; };
    float* xa0 = alloc((size_t)N_AP * HDIM);
    float* xu0 = alloc((size_t)N_USER * HDIM);
    float* xt0 = alloc((size_t)N_TARGET * HDIM);
    float* xa1 = alloc((size_t)N_AP * HDIM);
    float* xu1 = alloc((size_t)N_USER * HDIM);
    float* xt1 = alloc((size_t)N_TARGET * HDIM);
    float* m_user   = alloc((size_t)N_USER * HDIM);
    float* m_target = alloc((size_t)N_TARGET * HDIM);
    float* m_ap1 = alloc((size_t)N_AP * HDIM);
    float* m_ap2 = alloc((size_t)N_AP * HDIM);
    float* c_user   = alloc(N_USER);
    float* c_target = alloc(N_TARGET);
    float* c_ap1 = alloc(N_AP);
    float* c_ap2 = alloc(N_AP);

    // -------- projections to H=64 --------
    proj_kernel<<<(N_AP * HDIM + 255) / 256, 256, 0, stream>>>(x_ap, P(7), P(8), xa0, N_AP, 10);
    proj_kernel<<<(N_USER * HDIM + 255) / 256, 256, 0, stream>>>(x_user, P(9), P(10), xu0, N_USER, 8);
    proj_kernel<<<(N_TARGET * HDIM + 255) / 256, 256, 0, stream>>>(x_target, P(11), P(12), xt0, N_TARGET, 8);

    auto run_layer = [&](const float* xa, const float* xu, const float* xt,
                         float* ya, float* yu, float* yt, int pb) {
        // serves: AP -> USER mean
        hipMemsetAsync(m_user, 0, (size_t)N_USER * HDIM * sizeof(float), stream);
        hipMemsetAsync(c_user, 0, (size_t)N_USER * sizeof(float), stream);
        scatter_add_kernel<<<(E_SERVES * 16 + 255) / 256, 256, 0, stream>>>(
            xa, serves_src, serves_dst, m_user, c_user, E_SERVES);
        mean_div_kernel<<<(N_USER * HDIM + 255) / 256, 256, 0, stream>>>(m_user, c_user, N_USER);

        // senses: AP -> TARGET mean
        hipMemsetAsync(m_target, 0, (size_t)N_TARGET * HDIM * sizeof(float), stream);
        hipMemsetAsync(c_target, 0, (size_t)N_TARGET * sizeof(float), stream);
        scatter_add_kernel<<<(E_SENSES * 16 + 255) / 256, 256, 0, stream>>>(
            xa, senses_src, senses_dst, m_target, c_target, E_SENSES);
        mean_div_kernel<<<(N_TARGET * HDIM + 255) / 256, 256, 0, stream>>>(m_target, c_target, N_TARGET);

        // rev_serves: USER -> AP mean (LDS-replicated accumulator)
        hipMemsetAsync(m_ap1, 0, (size_t)N_AP * HDIM * sizeof(float), stream);
        hipMemsetAsync(c_ap1, 0, (size_t)N_AP * sizeof(float), stream);
        scatter_add_ap_kernel<<<256, 256, 0, stream>>>(xu, serves_dst, serves_src, m_ap1, c_ap1, E_SERVES);
        mean_div_kernel<<<(N_AP * HDIM + 255) / 256, 256, 0, stream>>>(m_ap1, c_ap1, N_AP);

        // rev_senses: TARGET -> AP mean
        hipMemsetAsync(m_ap2, 0, (size_t)N_AP * HDIM * sizeof(float), stream);
        hipMemsetAsync(c_ap2, 0, (size_t)N_AP * sizeof(float), stream);
        scatter_add_ap_kernel<<<256, 256, 0, stream>>>(xt, senses_dst, senses_src, m_ap2, c_ap2, E_SENSES);
        mean_div_kernel<<<(N_AP * HDIM + 255) / 256, 256, 0, stream>>>(m_ap2, c_ap2, N_AP);

        // dense combines (WMMA)
        sage_wmma_kernel<<<(N_USER + 63) / 64, 128, 0, stream>>>(
            m_user, xu, P(pb + 0), P(pb + 1), P(pb + 2), yu, N_USER);
        sage_wmma_kernel<<<(N_TARGET + 63) / 64, 128, 0, stream>>>(
            m_target, xt, P(pb + 3), P(pb + 4), P(pb + 5), yt, N_TARGET);
        ap_combine_wmma_kernel<<<(N_AP + 63) / 64, 128, 0, stream>>>(
            m_ap1, m_ap2, xa,
            P(pb + 6), P(pb + 7), P(pb + 8),
            P(pb + 9), P(pb + 10), P(pb + 11), ya, N_AP);
    };

    run_layer(xa0, xu0, xt0, xa1, xu1, xt1, 13);   // conv1
    run_layer(xa1, xu1, xt1, xa0, xu0, xt0, 25);   // conv2 (final feats back in buf0)

    // -------- heads --------
    float* out  = (float*)d_out;
    float* tau  = out;
    float* sres = tau + N_AP;
    float* xlog = sres + N_TARGET;
    float* ytx  = xlog + E_SERVES;
    float* yrx  = ytx + E_SENSES;

    node_head_kernel<<<(N_AP * 16 + 255) / 256, 256, 0, stream>>>(xa0, P(37), P(38), tau, N_AP);
    node_head_kernel<<<(N_TARGET * 16 + 255) / 256, 256, 0, stream>>>(xt0, P(39), P(40), sres, N_TARGET);
    edge_head1_kernel<<<(unsigned)(((long long)E_SERVES * 32 + 255) / 256), 256, 0, stream>>>(
        xa0, xu0, serves_src, serves_dst, P(41), P(42), xlog, E_SERVES);
    edge_head2_kernel<<<(unsigned)(((long long)E_SENSES * 32 + 255) / 256), 256, 0, stream>>>(
        xa0, xt0, senses_src, senses_dst, P(43), P(44), P(45), P(46), ytx, yrx, E_SENSES);
}
